// MultiHeadSelfAttention_10599979286772
// MI455X (gfx1250) — matrix operand order repaired and checked by reading, not compiled, not run on hardware
//
#include <hip/hip_runtime.h>

typedef _Float16 v16h __attribute__((ext_vector_type(16)));
typedef _Float16 v8h  __attribute__((ext_vector_type(8)));
typedef float    v8f  __attribute__((ext_vector_type(8)));

union H16 { v16h v; v8h h[2]; };

#define WMMA_F16(A, B, C) \
    __builtin_amdgcn_wmma_f32_16x16x32_f16(false, (A), false, (B), (short)0, (C), false, false)

constexpr int Bb = 8;
constexpr int Sq = 1024;
constexpr int Dm = 768;
constexpr int Hh = 12;
constexpr int HD = 64;

// ---------------------------------------------------------------------------
// Kernel 0: f32 -> f16 conversion (X once; each W once), 8 elems/thread.
// ---------------------------------------------------------------------------
__global__ void cvt_f16_kernel(const float* __restrict__ src,
                               _Float16* __restrict__ dst, int n8)
{
    const int i = blockIdx.x * blockDim.x + threadIdx.x;
    if (i >= n8) return;
    const float4 a = ((const float4*)src)[2 * i];
    const float4 b = ((const float4*)src)[2 * i + 1];
    v8h o;
    o[0] = (_Float16)a.x; o[1] = (_Float16)a.y;
    o[2] = (_Float16)a.z; o[3] = (_Float16)a.w;
    o[4] = (_Float16)b.x; o[5] = (_Float16)b.y;
    o[6] = (_Float16)b.z; o[7] = (_Float16)b.w;
    ((v8h*)dst)[i] = o;
}

// ---------------------------------------------------------------------------
// Kernel 1: QKV projection GEMM (f16 in, f32 acc).  Y[m,n] = sum_k X[m,k]W[n,k]
// Double-buffered LDS panels staged with GLOBAL_LOAD_ASYNC_TO_LDS_B128:
// panel k+1 DMA overlaps the WMMAs on panel k; async loads complete in order,
// so "s_wait_asynccnt 4" retires exactly the previous panel's 4 copies.
// Writes Q,K as [B,H,S,HD] f16 and V transposed as [B,H,HD,S] f16.
// ---------------------------------------------------------------------------
__global__ void qkv_gemm_kernel(const _Float16* __restrict__ Xh,
                                const _Float16* __restrict__ Wh,
                                _Float16* __restrict__ Qh,
                                _Float16* __restrict__ Kh,
                                _Float16* __restrict__ Vt)
{
    constexpr int LDT = 40;   // padded row stride in halves (80 B, 16B-aligned)
    __shared__ __attribute__((aligned(16))) _Float16 Xs[2][64 * LDT];
    __shared__ __attribute__((aligned(16))) _Float16 Ws[2][64 * LDT];

    const int tid  = threadIdx.x;
    const int lane = tid & 31;
    const int wave = tid >> 5;
    const int hi   = lane >> 4;
    const int ln   = lane & 15;

    const int m0    = blockIdx.x * 64;
    const int n0    = blockIdx.y * 64;
    const int which = blockIdx.z;
    const _Float16* __restrict__ Wsel = Wh + (size_t)which * Dm * Dm;

    v8f acc[4] = {};

    // async-copy chunk map: panel = 64 rows x 64 B = 256 chunks of 16 B
    const int c0 = tid, c1 = tid + 128;
    const int r0 = c0 >> 2, cc0 = (c0 & 3) * 16;   // byte offset in row
    const int r1 = c1 >> 2, cc1 = (c1 & 3) * 16;
    const unsigned off0 = (unsigned)(r0 * LDT * 2 + cc0);
    const unsigned off1 = (unsigned)(r1 * LDT * 2 + cc1);

    // issue the 4 async b128 copies for one (X,W) panel into buffer `buf`
    auto issue = [&](int buf, int k0) {
        const unsigned lX0 = (unsigned)(uintptr_t)(&Xs[buf][0]) + off0;
        const unsigned lX1 = (unsigned)(uintptr_t)(&Xs[buf][0]) + off1;
        const unsigned lW0 = (unsigned)(uintptr_t)(&Ws[buf][0]) + off0;
        const unsigned lW1 = (unsigned)(uintptr_t)(&Ws[buf][0]) + off1;
        const unsigned long long gX0 =
            (unsigned long long)(Xh + (size_t)(m0 + r0) * Dm + k0) + cc0;
        const unsigned long long gX1 =
            (unsigned long long)(Xh + (size_t)(m0 + r1) * Dm + k0) + cc1;
        const unsigned long long gW0 =
            (unsigned long long)(Wsel + (size_t)(n0 + r0) * Dm + k0) + cc0;
        const unsigned long long gW1 =
            (unsigned long long)(Wsel + (size_t)(n0 + r1) * Dm + k0) + cc1;
        asm volatile("global_load_async_to_lds_b128 %0, %1, off"
                     :: "v"(lX0), "v"(gX0) : "memory");
        asm volatile("global_load_async_to_lds_b128 %0, %1, off"
                     :: "v"(lX1), "v"(gX1) : "memory");
        asm volatile("global_load_async_to_lds_b128 %0, %1, off"
                     :: "v"(lW0), "v"(gW0) : "memory");
        asm volatile("global_load_async_to_lds_b128 %0, %1, off"
                     :: "v"(lW1), "v"(gW1) : "memory");
    };

    issue(0, 0);                                   // prologue: panel 0

    for (int k0 = 0; k0 < Dm; k0 += 32) {
        const int cur = (k0 >> 5) & 1;
        if (k0 + 32 < Dm) {
            issue(cur ^ 1, k0 + 32);               // prefetch next panel (DMA)
            // 4 newest copies may stay in flight; previous panel retired:
            asm volatile("s_wait_asynccnt 0x4" ::: "memory");
        } else {
            asm volatile("s_wait_asynccnt 0x0" ::: "memory");
        }
        __syncthreads();

        // ---- A fragment: 16x32, lane=M row, halves k split 8+8 ----
        H16 a;
        const _Float16* ap = &Xs[cur][(wave * 16 + ln) * LDT + hi * 8];
        a.h[0] = *(const v8h*)ap;
        a.h[1] = *(const v8h*)(ap + 16);

#pragma unroll
        for (int nt = 0; nt < 4; ++nt) {           // B: lane=N col, halves k split 8+8
            H16 b;
            const _Float16* bp = &Ws[cur][(nt * 16 + ln) * LDT + hi * 8];
            b.h[0] = *(const v8h*)bp;
            b.h[1] = *(const v8h*)(bp + 16);
            acc[nt] = WMMA_F16(a.v, b.v, acc[nt]);
        }
        __syncthreads();
    }

    // ---- epilogue: b and h are block constants; branch hoisted ----
    const int btok = m0 >> 10;                     // 1024-token batches, 64|1024
    const int hout = n0 >> 6;                      // HD=64 aligned N tiles
    const int srow = (m0 & (Sq - 1)) + wave * 16 + 8 * hi;   // + r

    if (which == 2) {
        // V transposed [B,H,HD,S]: r is contiguous -> packed v8h stores
        _Float16* base = Vt + (((size_t)btok * Hh + hout) * HD) * Sq + srow;
#pragma unroll
        for (int nt = 0; nt < 4; ++nt) {
            const int d = nt * 16 + ln;
            v8h pkd;
#pragma unroll
            for (int r = 0; r < 8; ++r) pkd[r] = (_Float16)acc[nt][r];
            *(v8h*)(base + (size_t)d * Sq) = pkd;
        }
    } else {
        _Float16* base = ((which == 0) ? Qh : Kh) +
                         (((size_t)btok * Hh + hout) * Sq + srow) * HD + ln;
#pragma unroll
        for (int nt = 0; nt < 4; ++nt)
#pragma unroll
            for (int r = 0; r < 8; ++r)
                base[(size_t)r * HD + nt * 16] = (_Float16)acc[nt][r];
    }
}

// ---------------------------------------------------------------------------
// Kernel 2: flash attention over one (b,h). Each wave owns 16 query rows.
// S^T tiles (keys in regs, query in lane) -> softmax key-reduction is
// in-register + one xor-16 shuffle; O^T accumulated via WMMA.
// ---------------------------------------------------------------------------
__global__ void attn_kernel(const _Float16* __restrict__ Qh,
                            const _Float16* __restrict__ Kh,
                            const _Float16* __restrict__ Vt,
                            float* __restrict__ Attn)
{
    const int tid  = threadIdx.x;
    const int lane = tid & 31;
    const int wave = tid >> 5;
    const int hi   = lane >> 4;
    const int ln   = lane & 15;

    const int b  = blockIdx.z;
    const int h  = blockIdx.y;
    const int q0 = blockIdx.x * 64 + wave * 16;

    const _Float16* __restrict__ Qp = Qh + ((size_t)b * Hh + h) * Sq * HD;
    const _Float16* __restrict__ Kp = Kh + ((size_t)b * Hh + h) * Sq * HD;
    const _Float16* __restrict__ Vp = Vt + ((size_t)b * Hh + h) * HD * Sq;

    // Q^T B-fragments (loop invariant): B[d][q], lane=q, halves d split 8+8
    H16 bq0, bq1;
    {
        const _Float16* p = Qp + (size_t)(q0 + ln) * HD + hi * 8;
        bq0.h[0] = *(const v8h*)p;        bq0.h[1] = *(const v8h*)(p + 16);
        bq1.h[0] = *(const v8h*)(p + 32); bq1.h[1] = *(const v8h*)(p + 48);
    }

    v8f o[4] = {};
    float m_run = -INFINITY;
    float l_run = 0.0f;

    for (int kb = 0; kb < Sq; kb += 32) {
        const int pk = (kb + 32 < Sq) ? kb + 32 : 0;
        __builtin_prefetch(Kp + (size_t)(pk + ln) * HD, 0, 1);

        // ---- S^T = K(A) x Q^T(B), two 16-key tiles ----
        v8f s0 = {}, s1 = {};
        {
            const _Float16* p = Kp + (size_t)(kb + ln) * HD + hi * 8;
            H16 a0, a1;
            a0.h[0] = *(const v8h*)p;        a0.h[1] = *(const v8h*)(p + 16);
            a1.h[0] = *(const v8h*)(p + 32); a1.h[1] = *(const v8h*)(p + 48);
            s0 = WMMA_F16(a0.v, bq0.v, s0);
            s0 = WMMA_F16(a1.v, bq1.v, s0);
        }
        {
            const _Float16* p = Kp + (size_t)(kb + 16 + ln) * HD + hi * 8;
            H16 a0, a1;
            a0.h[0] = *(const v8h*)p;        a0.h[1] = *(const v8h*)(p + 16);
            a1.h[0] = *(const v8h*)(p + 32); a1.h[1] = *(const v8h*)(p + 48);
            s1 = WMMA_F16(a0.v, bq0.v, s1);
            s1 = WMMA_F16(a1.v, bq1.v, s1);
        }

        // ---- online softmax over 32 keys ----
        float tm = -INFINITY;
#pragma unroll
        for (int r = 0; r < 8; ++r) {
            s0[r] *= 0.125f;  s1[r] *= 0.125f;   // 1/sqrt(64)
            tm = fmaxf(tm, fmaxf(s0[r], s1[r]));
        }
        tm = fmaxf(tm, __shfl_xor(tm, 16, 32));
        const float m_new = fmaxf(m_run, tm);
        const float alpha = __expf(m_run - m_new);

        float rs = 0.0f;
#pragma unroll
        for (int r = 0; r < 8; ++r) {
            s0[r] = __expf(s0[r] - m_new);
            s1[r] = __expf(s1[r] - m_new);
            rs += s0[r] + s1[r];
        }
        rs += __shfl_xor(rs, 16, 32);
        l_run = l_run * alpha + rs;
        m_run = m_new;

#pragma unroll
        for (int dt = 0; dt < 4; ++dt)
#pragma unroll
            for (int r = 0; r < 8; ++r) o[dt][r] *= alpha;

        // ---- build P^T B-fragment (32 keys x 16 q) via xor-16 exchange ----
        H16 bp_;
#pragma unroll
        for (int r = 0; r < 8; ++r) {
            const float e0 = __shfl_xor(s0[r], 16, 32);
            const float e1 = __shfl_xor(s1[r], 16, 32);
            bp_.h[0][r] = (_Float16)(hi ? e1 : s0[r]);   // keys hi*16 + 0..7
            bp_.h[1][r] = (_Float16)(hi ? s1[r] : e0);   // keys hi*16 + 8..15
        }

        // ---- O^T += V^T(A) x P^T(B) over four d-tiles ----
#pragma unroll
        for (int dt = 0; dt < 4; ++dt) {
            const _Float16* p = Vp + (size_t)(dt * 16 + ln) * Sq + kb + hi * 16;
            H16 av;
            av.h[0] = *(const v8h*)p;
            av.h[1] = *(const v8h*)(p + 8);
            o[dt] = WMMA_F16(av.v, bp_.v, o[dt]);
        }
    }

    // ---- normalize + packed float4 stores (r contiguous in d) ----
    const float inv = 1.0f / l_run;
    float* obase = Attn + ((size_t)b * Sq + q0 + ln) * Dm + h * HD + 8 * hi;
#pragma unroll
    for (int dt = 0; dt < 4; ++dt) {
        const float4 lo = make_float4(o[dt][0] * inv, o[dt][1] * inv,
                                      o[dt][2] * inv, o[dt][3] * inv);
        const float4 hi4 = make_float4(o[dt][4] * inv, o[dt][5] * inv,
                                       o[dt][6] * inv, o[dt][7] * inv);
        *(float4*)(obase + dt * 16)     = lo;
        *(float4*)(obase + dt * 16 + 4) = hi4;
    }
}

// ---------------------------------------------------------------------------
// Kernel 3: residual add + LayerNorm + ReLU. One token per 256-thread block.
// ---------------------------------------------------------------------------
__global__ void ln_relu_kernel(const float* __restrict__ Attn,
                               const float* __restrict__ X,
                               const float* __restrict__ gamma,
                               const float* __restrict__ beta,
                               float* __restrict__ Out)
{
    __shared__ float redbuf[8];
    const int t    = blockIdx.x;
    const int lane = threadIdx.x & 31;
    const int wrp  = threadIdx.x >> 5;

    const float* a = Attn + (size_t)t * Dm;
    const float* x = X    + (size_t)t * Dm;

    float v[3];
    float sum = 0.0f;
#pragma unroll
    for (int i = 0; i < 3; ++i) {
        const int idx = threadIdx.x + i * 256;
        v[i] = a[idx] + x[idx];
        sum += v[i];
    }
#pragma unroll
    for (int off = 16; off; off >>= 1) sum += __shfl_xor(sum, off, 32);
    if (lane == 0) redbuf[wrp] = sum;
    __syncthreads();
    float tot = 0.0f;
#pragma unroll
    for (int w = 0; w < 8; ++w) tot += redbuf[w];
    const float mu = tot * (1.0f / (float)Dm);
    __syncthreads();

    float sq = 0.0f;
#pragma unroll
    for (int i = 0; i < 3; ++i) { const float d = v[i] - mu; sq += d * d; }
#pragma unroll
    for (int off = 16; off; off >>= 1) sq += __shfl_xor(sq, off, 32);
    if (lane == 0) redbuf[wrp] = sq;
    __syncthreads();
    float tot2 = 0.0f;
#pragma unroll
    for (int w = 0; w < 8; ++w) tot2 += redbuf[w];
    const float inv = rsqrtf(tot2 * (1.0f / (float)Dm) + 1e-5f);

#pragma unroll
    for (int i = 0; i < 3; ++i) {
        const int idx = threadIdx.x + i * 256;
        const float y = (v[i] - mu) * inv * gamma[idx] + beta[idx];
        Out[(size_t)t * Dm + idx] = fmaxf(y, 0.0f);
    }
}

// ---------------------------------------------------------------------------
extern "C" void kernel_launch(void* const* d_in, const int* in_sizes, int n_in,
                              void* d_out, int out_size, void* d_ws, size_t ws_size,
                              hipStream_t stream)
{
    const float* X     = (const float*)d_in[0];
    const float* Wq    = (const float*)d_in[1];
    const float* Wk    = (const float*)d_in[2];
    const float* Wv    = (const float*)d_in[3];
    const float* gamma = (const float*)d_in[4];
    const float* beta  = (const float*)d_in[5];
    float*       out   = (float*)d_out;

    const size_t tokElems = (size_t)Bb * Sq * Dm;   // 6,291,456
    const size_t wElems   = (size_t)Dm * Dm;        //   589,824
    char* ws = (char*)d_ws;
    _Float16* Qh   = (_Float16*)ws;
    _Float16* Kh   = Qh + tokElems;
    _Float16* Vt   = Kh + tokElems;
    _Float16* Xh   = Vt + tokElems;
    _Float16* Wh   = Xh + tokElems;                 // [3][768][768]
    float*    Attn = (float*)((char*)(Wh + 3 * wElems));

    // f32 -> f16 pre-pass
    cvt_f16_kernel<<<(int)(tokElems / 8 + 255) / 256, 256, 0, stream>>>(X, Xh, (int)(tokElems / 8));
    cvt_f16_kernel<<<(int)(wElems / 8 + 255) / 256, 256, 0, stream>>>(Wq, Wh,              (int)(wElems / 8));
    cvt_f16_kernel<<<(int)(wElems / 8 + 255) / 256, 256, 0, stream>>>(Wk, Wh + wElems,     (int)(wElems / 8));
    cvt_f16_kernel<<<(int)(wElems / 8 + 255) / 256, 256, 0, stream>>>(Wv, Wh + 2 * wElems, (int)(wElems / 8));

    qkv_gemm_kernel<<<dim3(Bb * Sq / 64, Dm / 64, 3), 128, 0, stream>>>(Xh, Wh, Qh, Kh, Vt);

    attn_kernel<<<dim3(Sq / 64, Hh, Bb), 128, 0, stream>>>(Qh, Kh, Vt, Attn);

    ln_relu_kernel<<<Bb * Sq, 256, 0, stream>>>(Attn, X, gamma, beta, out);
}
